// SubconsciousPlanning_10496900071805
// MI455X (gfx1250) — matrix operand order repaired and checked by reading, not compiled, not run on hardware
//
#include <hip/hip_runtime.h>
#include <hip/hip_bf16.h>
#include <math.h>

// ---------------------------------------------------------------------------
// Problem constants (from reference)
// ---------------------------------------------------------------------------
#define BATCH   4096
#define HIDDEN  4096
#define NOISE   2048
#define NUM_GEN 5
#define TOPK    8
#define EVAL_H  64

typedef __attribute__((ext_vector_type(16))) __bf16 v16bf;
typedef __attribute__((ext_vector_type(2)))  __bf16 v2bf;
typedef __attribute__((ext_vector_type(8)))  float  v8f;

// ---------------------------------------------------------------------------
// bf16 helpers
// ---------------------------------------------------------------------------
__device__ __forceinline__ unsigned short f2bf(float f) {
    unsigned int u = __builtin_bit_cast(unsigned int, f);
    u += 0x7fffu + ((u >> 16) & 1u);
    return (unsigned short)(u >> 16);
}
__device__ __forceinline__ float bf2f(unsigned short s) {
    unsigned int u = ((unsigned int)s) << 16;
    return __builtin_bit_cast(float, u);
}
// packed f32x2 -> bf16x2 (hardware v_cvt_pk_bf16_f32 when available)
__device__ __forceinline__ unsigned int pack2bf(float a, float b) {
#if __has_builtin(__builtin_amdgcn_cvt_pk_bf16_f32)
    v2bf r = __builtin_amdgcn_cvt_pk_bf16_f32(a, b);
    return __builtin_bit_cast(unsigned int, r);
#else
    return (unsigned int)f2bf(a) | ((unsigned int)f2bf(b) << 16);
#endif
}
__device__ __forceinline__ float gelu_exact(float x) {
    return 0.5f * x * (1.0f + erff(x * 0.70710678118654752f));
}

struct U32x8 { uint4 lo, hi; };   // 32B container -> bit_cast to v16bf

// ---------------------------------------------------------------------------
// WMMA GEMM: C[M,N] = act(A[M,K] @ B[K,N] + bias [+ aux*scale])
//   128x128 block tile, BK=32, 8 waves (2x4), wave tile 64x32
//   Double-buffered LDS, register staging, f32->bf16 convert on store.
//   A in LDS: As[buf][m][k]  (pitch 40 ushorts -> 16B-aligned frag chunks)
//   B in LDS: BsT[buf][n][k] (transposed; staged as 2x2 blocks so the
//             transpose store is one packed b32 per column pair)
// Epilogues:
//   0: out_f32 = acc + bias
//   1: t = tanh(acc + bias + aux*scale); out_f32 = t; out_bf16 = t
//   2: out_bf16 = acc + bias        (z-batched)
//   3: out_f32 = tanh(acc + bias) * scale
// ---------------------------------------------------------------------------
template<int EPI, bool ABF16>
__global__ __launch_bounds__(256)
void gemm_wmma_kernel(const void* __restrict__ Aall, size_t strideAz, int ldA,
                      const float* __restrict__ Ball, size_t strideBz, int ldB,
                      const float* __restrict__ biasAll, int strideBiasZ,
                      const float* __restrict__ aux, int ldAux,
                      const float* __restrict__ scalePtr,
                      float* __restrict__ outF32,
                      unsigned short* __restrict__ outBF16, size_t strideOz,
                      int ldO, int K)
{
    __shared__ unsigned short As [2][128][40];   // [buf][M][K] bf16
    __shared__ unsigned short BsT[2][128][40];   // [buf][N][K] bf16 (transposed)

    const int tid  = threadIdx.x;
    const int lane = tid & 31;
    const int wave = tid >> 5;
    const int half = lane >> 4;
    const int l16  = lane & 15;
    const int wm   = (wave >> 2) * 64;
    const int wn   = (wave & 3) * 32;

    const int m0 = blockIdx.y * 128;
    const int n0 = blockIdx.x * 128;
    const int z  = blockIdx.z;

    const float*          Af = (const float*)Aall          + (ABF16 ? 0 : (size_t)z * strideAz);
    const unsigned short* Ah = (const unsigned short*)Aall + (ABF16 ? (size_t)z * strideAz : 0);
    const float*          B  = Ball    + (size_t)z * strideBz;
    const float*          bs = biasAll + (size_t)z * strideBiasZ;

    float scl = 0.0f;
    if (EPI == 1 || EPI == 3) scl = scalePtr[0];

    v8f acc[4][2] = {};

    // staging registers (one K-tile in flight)
    uint2  ha[4];          // A as raw bf16 pairs (ABF16 path)
    float4 fa[4];          // A as f32 (convert path)
    float2 ub[4], vb[4];   // B rows k and k+1 (2x2 block staging)

    // B staging coordinates: thread -> 4 blocks of 2 rows x 2 cols
    const int bc = (tid & 63) << 1;        // column pair base (0..126)
    const int bkbase = (tid >> 6);         // 0..3

    const int KT = K >> 5;

    // ---- tile load into registers (global, fully coalesced) ----
    auto load_tile = [&](int kt) {
        const int k0 = kt << 5;
        #pragma unroll
        for (int p = 0; p < 4; ++p) {
            int l = (tid << 2) + (p << 10);
            int r = l >> 5, c = l & 31;
            if (ABF16)
                ha[p] = *reinterpret_cast<const uint2*>(Ah + (size_t)(m0 + r) * ldA + k0 + c);
            else
                fa[p] = *reinterpret_cast<const float4*>(Af + (size_t)(m0 + r) * ldA + k0 + c);
        }
        #pragma unroll
        for (int p = 0; p < 4; ++p) {
            int k = (bkbase + (p << 2)) << 1;   // even row of the 2x2 block
            ub[p] = *reinterpret_cast<const float2*>(B + (size_t)(k0 + k)     * ldB + n0 + bc);
            vb[p] = *reinterpret_cast<const float2*>(B + (size_t)(k0 + k + 1) * ldB + n0 + bc);
        }
    };
    // ---- register -> LDS (convert to bf16; B transposed via 2x2 blocks) ----
    auto store_tile = [&](int buf) {
        #pragma unroll
        for (int p = 0; p < 4; ++p) {
            int l = (tid << 2) + (p << 10);
            int r = l >> 5, c = l & 31;
            uint2 q;
            if (ABF16) q = ha[p];
            else { q.x = pack2bf(fa[p].x, fa[p].y); q.y = pack2bf(fa[p].z, fa[p].w); }
            *reinterpret_cast<uint2*>(&As[buf][r][c]) = q;
        }
        #pragma unroll
        for (int p = 0; p < 4; ++p) {
            int k = (bkbase + (p << 2)) << 1;
            // (k,k+1) contiguous in BsT[n][.] -> single packed b32 store per col
            *reinterpret_cast<unsigned int*>(&BsT[buf][bc][k])     = pack2bf(ub[p].x, vb[p].x);
            *reinterpret_cast<unsigned int*>(&BsT[buf][bc + 1][k]) = pack2bf(ub[p].y, vb[p].y);
        }
    };

    // prologue: stage tile 0
    load_tile(0);
    store_tile(0);
    __syncthreads();

    for (int kt = 0; kt < KT; ++kt) {
        const int cur = kt & 1;
        if (kt + 1 < KT) load_tile(kt + 1);         // issue next-tile loads early
        if (kt + 2 < KT) {                           // L2 prefetch of tile kt+2
            const int k2 = (kt + 2) << 5;
            __builtin_prefetch(B + (size_t)(k2 + (tid >> 3)) * ldB + n0 + ((tid & 7) << 4), 0, 0);
            if (ABF16)
                __builtin_prefetch(Ah + (size_t)(m0 + (tid >> 1)) * ldA + k2 + ((tid & 1) << 4), 0, 0);
            else
                __builtin_prefetch(Af + (size_t)(m0 + (tid >> 1)) * ldA + k2 + ((tid & 1) << 4), 0, 0);
        }

        // ---- fragments (contiguous 16B LDS chunks -> ds_load_b128) ----
        v16bf afr[4];
        #pragma unroll
        for (int mi = 0; mi < 4; ++mi) {
            const unsigned short* p = &As[cur][wm + mi * 16 + l16][0];
            U32x8 t;
            t.lo = *reinterpret_cast<const uint4*>(p + 16 * half);       // k = 16h..16h+7
            t.hi = *reinterpret_cast<const uint4*>(p + 16 * half + 8);   // k = 16h+8..16h+15
            afr[mi] = __builtin_bit_cast(v16bf, t);
        }
        v16bf bfr[2];
        #pragma unroll
        for (int ni = 0; ni < 2; ++ni) {
            const unsigned short* p = &BsT[cur][wn + ni * 16 + l16][16 * half]; // k = 16h..16h+15
            U32x8 t;
            t.lo = *reinterpret_cast<const uint4*>(p);
            t.hi = *reinterpret_cast<const uint4*>(p + 8);
            bfr[ni] = __builtin_bit_cast(v16bf, t);
        }
        // ---- 8 WMMAs ----
        #pragma unroll
        for (int mi = 0; mi < 4; ++mi)
            #pragma unroll
            for (int ni = 0; ni < 2; ++ni)
                acc[mi][ni] = __builtin_amdgcn_wmma_f32_16x16x32_bf16(
                    false, afr[mi], false, bfr[ni], (short)0, acc[mi][ni],
                    false, false);

        if (kt + 1 < KT) {
            store_tile(1 - cur);     // other buffer: no race with current reads
            __syncthreads();         // publish for next iteration
        }
    }

    // ---- epilogue: C/D layout -> lanes 0-15 M=v, lanes 16-31 M=8+v ----
    #pragma unroll
    for (int mi = 0; mi < 4; ++mi) {
        #pragma unroll
        for (int ni = 0; ni < 2; ++ni) {
            int tm = m0 + wm + mi * 16 + half * 8;
            int n  = n0 + wn + ni * 16 + l16;
            float bn = bs[n];
            #pragma unroll
            for (int v = 0; v < 8; ++v) {
                int m = tm + v;
                float x = acc[mi][ni][v] + bn;
                size_t o = (size_t)m * ldO + n;
                if (EPI == 0) {
                    outF32[o] = x;
                } else if (EPI == 1) {
                    x += aux[(size_t)m * ldAux + n] * scl;
                    float t = tanhf(x);
                    outF32[o]  = t;
                    outBF16[o] = (unsigned short)pack2bf(t, t);
                } else if (EPI == 2) {
                    outBF16[(size_t)z * strideOz + o] = (unsigned short)pack2bf(x, x);
                } else { // EPI == 3
                    outF32[o] = tanhf(x) * scl;
                }
            }
        }
    }
}

// ---------------------------------------------------------------------------
// Top-8 per row (wave32: one wave per batch row)
// ---------------------------------------------------------------------------
__global__ __launch_bounds__(256)
void top8_kernel(const float* __restrict__ scores,
                 const float* __restrict__ field,
                 int* __restrict__ idx_out, float* __restrict__ val_out)
{
    const int wave = threadIdx.x >> 5;
    const int lane = threadIdx.x & 31;
    const int b    = blockIdx.x * 8 + wave;
    const float* S = scores + (size_t)b * NOISE;

    int picked[TOPK];
    for (int it = 0; it < TOPK; ++it) {
        float bm = -INFINITY;
        int   bi = 0x7fffffff;
        for (int j = lane; j < NOISE; j += 32) {
            bool skip = false;
            for (int t = 0; t < it; ++t) skip |= (picked[t] == j);
            if (skip) continue;
            float v = S[j];
            if (v > bm || (v == bm && j < bi)) { bm = v; bi = j; }
        }
        #pragma unroll
        for (int off = 16; off; off >>= 1) {
            float om = __shfl_xor(bm, off, 32);
            int   oi = __shfl_xor(bi, off, 32);
            if (om > bm || (om == bm && oi < bi)) { bm = om; bi = oi; }
        }
        picked[it] = bi;
    }
    if (lane < TOPK) {
        idx_out[b * TOPK + lane] = picked[lane];
        val_out[b * TOPK + lane] = field[(size_t)b * NOISE + picked[lane]];
    }
}

// ---------------------------------------------------------------------------
// Layer 2 with top-8 sparsity: h[g][b][:] = gelu(sum_j val_j*Wg1[g,idx_j,:]+bg1[g])
// ---------------------------------------------------------------------------
__global__ __launch_bounds__(256)
void sparse_gen_kernel(const int* __restrict__ idx, const float* __restrict__ val,
                       const float* __restrict__ Wg1, const float* __restrict__ bg1,
                       unsigned short* __restrict__ h)
{
    const int b = blockIdx.x;
    const int g = blockIdx.y;
    int   id[TOPK];
    float vl[TOPK];
    #pragma unroll
    for (int j = 0; j < TOPK; ++j) {
        id[j] = idx[b * TOPK + j];
        vl[j] = val[b * TOPK + j];
    }
    const float* bg    = bg1 + (size_t)g * HIDDEN;
    const size_t wbase = (size_t)g * NOISE * HIDDEN;
    unsigned short* hp = h + ((size_t)g * BATCH + b) * HIDDEN;

    for (int n = threadIdx.x; n < HIDDEN; n += 256) {
        float a = bg[n];
        #pragma unroll
        for (int j = 0; j < TOPK; ++j)
            a += vl[j] * Wg1[wbase + (size_t)id[j] * HIDDEN + n];
        float gx = gelu_exact(a);
        hp[n] = (unsigned short)pack2bf(gx, gx);
    }
}

// ---------------------------------------------------------------------------
// Evaluator: sc[b,g] = gelu(futures[g,b,:] @ We1 + be1) @ We2 + be2
// ---------------------------------------------------------------------------
__global__ __launch_bounds__(256)
void eval_kernel(const unsigned short* __restrict__ futures,
                 const float* __restrict__ We1, const float* __restrict__ be1,
                 const float* __restrict__ We2, const float* __restrict__ be2,
                 float* __restrict__ sc)
{
    const int b = blockIdx.x;
    const int g = blockIdx.y;
    const int wave = threadIdx.x >> 5;
    const int lane = threadIdx.x & 31;
    __shared__ float e[EVAL_H];

    const unsigned short* F = futures + ((size_t)g * BATCH + b) * HIDDEN;
    const int nb = wave * 8;

    float a[8] = {0.f, 0.f, 0.f, 0.f, 0.f, 0.f, 0.f, 0.f};
    for (int k = lane; k < HIDDEN; k += 32) {
        float f = bf2f(F[k]);
        const float* wr = We1 + (size_t)k * EVAL_H + nb;
        #pragma unroll
        for (int t = 0; t < 8; ++t) a[t] += f * wr[t];
    }
    #pragma unroll
    for (int t = 0; t < 8; ++t)
        #pragma unroll
        for (int off = 16; off; off >>= 1)
            a[t] += __shfl_xor(a[t], off, 32);
    if (lane == 0)
        #pragma unroll
        for (int t = 0; t < 8; ++t)
            e[nb + t] = gelu_exact(a[t] + be1[nb + t]);
    __syncthreads();
    if (threadIdx.x == 0) {
        float s = be2[0];
        #pragma unroll
        for (int n = 0; n < EVAL_H; ++n) s += e[n] * We2[n];
        sc[(size_t)b * NUM_GEN + g] = s;
    }
}

// ---------------------------------------------------------------------------
// Per-row argmax over 5 generators + gather best future row (bf16)
// ---------------------------------------------------------------------------
__global__ __launch_bounds__(256)
void select_kernel(const float* __restrict__ sc,
                   const unsigned short* __restrict__ futures,
                   unsigned short* __restrict__ best)
{
    const int b = blockIdx.x;
    __shared__ int bestg;
    if (threadIdx.x == 0) {
        float m = sc[(size_t)b * NUM_GEN];
        int bi = 0;
        #pragma unroll
        for (int g = 1; g < NUM_GEN; ++g) {
            float v = sc[(size_t)b * NUM_GEN + g];
            if (v > m) { m = v; bi = g; }
        }
        bestg = bi;
    }
    __syncthreads();
    const unsigned short* F = futures + ((size_t)bestg * BATCH + b) * HIDDEN;
    unsigned short* O = best + (size_t)b * HIDDEN;
    for (int n = threadIdx.x; n < HIDDEN; n += 256) O[n] = F[n];
}

// ---------------------------------------------------------------------------
// Launcher
// ---------------------------------------------------------------------------
extern "C" void kernel_launch(void* const* d_in, const int* in_sizes, int n_in,
                              void* d_out, int out_size, void* d_ws, size_t ws_size,
                              hipStream_t stream)
{
    (void)in_sizes; (void)n_in; (void)out_size; (void)ws_size;

    const float* context   = (const float*)d_in[0];   // [B,H]
    const float* noise     = (const float*)d_in[1];   // [B,NOISE]
    const float* W_proj    = (const float*)d_in[2];   // [H,NOISE]
    const float* b_proj    = (const float*)d_in[3];
    const float* noise_s   = (const float*)d_in[4];   // scalar (device)
    const float* W_sel     = (const float*)d_in[5];   // [NOISE,NOISE]
    const float* b_sel     = (const float*)d_in[6];
    const float* Wg1       = (const float*)d_in[7];   // [5,NOISE,H]
    const float* bg1       = (const float*)d_in[8];   // [5,H]
    const float* Wg2       = (const float*)d_in[9];   // [5,H,H]
    const float* bg2       = (const float*)d_in[10];  // [5,H]
    const float* We1       = (const float*)d_in[11];  // [H,64]
    const float* be1       = (const float*)d_in[12];
    const float* We2       = (const float*)d_in[13];  // [64,1]
    const float* be2       = (const float*)d_in[14];
    const float* W_inf     = (const float*)d_in[15];  // [H,H]
    const float* b_inf     = (const float*)d_in[16];
    const float* infl_s    = (const float*)d_in[17];  // scalar (device)
    float* out             = (float*)d_out;           // [B,H] f32

    // ---- workspace layout (bytes) ----
    char* w = (char*)d_ws;
    const size_t OFF_FIELD    = 0;                                   // f32 [B,NOISE]
    const size_t OFF_FIELD_BF = OFF_FIELD    + (size_t)BATCH*NOISE*4;// bf16 [B,NOISE]
    const size_t OFF_IDX      = OFF_FIELD_BF + (size_t)BATCH*NOISE*2;
    const size_t OFF_VAL      = OFF_IDX      + (size_t)BATCH*TOPK*4;
    const size_t OFF_SC       = OFF_VAL      + (size_t)BATCH*TOPK*4;
    const size_t OFF_BEST     = OFF_SC       + (size_t)BATCH*NUM_GEN*4; // bf16 [B,H]
    const size_t OFF_SCORES   = OFF_BEST     + (size_t)BATCH*HIDDEN*2;  // f32 [B,NOISE] (aliased under H)
    const size_t OFF_H        = OFF_SCORES;                             // bf16 [5,B,H] (overwrites scores)
    const size_t OFF_FUT      = OFF_H        + (size_t)NUM_GEN*BATCH*HIDDEN*2; // bf16 [5,B,H]

    float*          field_f  = (float*)(w + OFF_FIELD);
    unsigned short* field_bf = (unsigned short*)(w + OFF_FIELD_BF);
    int*            idx      = (int*)(w + OFF_IDX);
    float*          val      = (float*)(w + OFF_VAL);
    float*          sc       = (float*)(w + OFF_SC);
    unsigned short* best     = (unsigned short*)(w + OFF_BEST);
    float*          scores   = (float*)(w + OFF_SCORES);
    unsigned short* h        = (unsigned short*)(w + OFF_H);
    unsigned short* fut      = (unsigned short*)(w + OFF_FUT);

    const dim3 blk(256);

    // K1: field = tanh(context @ W_proj + b_proj + noise*noise_scale) -> f32 + bf16
    gemm_wmma_kernel<1, false><<<dim3(NOISE/128, BATCH/128, 1), blk, 0, stream>>>(
        (const void*)context, 0, HIDDEN, W_proj, 0, NOISE, b_proj, 0,
        noise, NOISE, noise_s, field_f, field_bf, 0, NOISE, HIDDEN);

    // K2: scores = field @ W_sel + b_sel -> f32
    gemm_wmma_kernel<0, true><<<dim3(NOISE/128, BATCH/128, 1), blk, 0, stream>>>(
        (const void*)field_bf, 0, NOISE, W_sel, 0, NOISE, b_sel, 0,
        nullptr, 0, nullptr, scores, nullptr, 0, NOISE, NOISE);

    // K3: top-8 per row -> idx, val
    top8_kernel<<<dim3(BATCH/8), blk, 0, stream>>>(scores, field_f, idx, val);

    // K4: h[g,b,:] = gelu(sparse ctx @ Wg1[g] + bg1[g]) -> bf16
    sparse_gen_kernel<<<dim3(BATCH, NUM_GEN), blk, 0, stream>>>(idx, val, Wg1, bg1, h);

    // K5: futures[g] = h[g] @ Wg2[g] + bg2[g] -> bf16 (batched over z=5)
    gemm_wmma_kernel<2, true><<<dim3(HIDDEN/128, BATCH/128, NUM_GEN), blk, 0, stream>>>(
        (const void*)h, (size_t)BATCH*HIDDEN, HIDDEN,
        Wg2, (size_t)HIDDEN*HIDDEN, HIDDEN,
        bg2, HIDDEN, nullptr, 0, nullptr,
        nullptr, fut, (size_t)BATCH*HIDDEN, HIDDEN, HIDDEN);

    // K6: sc[b,g] = gelu(futures @ We1 + be1) @ We2 + be2
    eval_kernel<<<dim3(BATCH, NUM_GEN), blk, 0, stream>>>(fut, We1, be1, We2, be2, sc);

    // K6b: best[b,:] = futures[argmax_g sc[b,g], b, :]
    select_kernel<<<dim3(BATCH), blk, 0, stream>>>(sc, fut, best);

    // K7: influence = tanh(best @ W_inf + b_inf) * influence_scale -> f32 out
    gemm_wmma_kernel<3, true><<<dim3(HIDDEN/128, BATCH/128, 1), blk, 0, stream>>>(
        (const void*)best, 0, HIDDEN, W_inf, 0, HIDDEN, b_inf, 0,
        nullptr, 0, infl_s, out, nullptr, 0, HIDDEN, HIDDEN);
}
